// RGLRU_88639535055094
// MI455X (gfx1250) — matrix operand order repaired and checked by reading, not compiled, not run on hardware
//
#include <hip/hip_runtime.h>
#include <cmath>

// RG-LRU forward for MI455X (gfx1250, wave32).
// Pipeline: K1 (gates via bf16 WMMA + local chunk scan -> summaries)
//           K2 (carry scan over chunk summaries)
//           K3 (gates via bf16 WMMA + scan seeded with carry -> y)
// Workspace: 3 * B*C*D floats (~12 MB).

#define H_    8
#define BW    128
#define D_    1024
#define CHUNK 32

typedef __attribute__((ext_vector_type(16))) __bf16 v16bf;
typedef __attribute__((ext_vector_type(8)))  __bf16 v8bf;
typedef __attribute__((ext_vector_type(8)))  float  v8f;

union FragBF { v8bf h2[2]; v16bf v; };

#if defined(__has_builtin)
#if __has_builtin(__builtin_amdgcn_global_load_async_to_lds_b128)
#define HAVE_ASYNC_LDS 1
#endif
#endif

#ifdef HAVE_ASYNC_LDS
// Builtin parameter types per hipcc diagnostic:
//   param 1: int __attribute__((vector_size(16))) __device__* (AS1, global src)
//   param 2: LDS destination (AS3)
typedef int v4i_t __attribute__((vector_size(16)));
typedef __attribute__((address_space(1))) v4i_t* gbl_v4i_p;
typedef __attribute__((address_space(3))) v4i_t* lds_v4i_p;
#endif

__device__ __forceinline__ void g2l_copy16(void* ldst, const void* gsrc) {
#ifdef HAVE_ASYNC_LDS
  // CDNA5 async global->LDS DMA (ASYNCcnt tracked), bypasses VGPRs.
  __builtin_amdgcn_global_load_async_to_lds_b128(
      (gbl_v4i_p)gsrc, (lds_v4i_p)ldst, 0, 0);
#else
  *(float4*)ldst = *(const float4*)gsrc;
#endif
}

__device__ __forceinline__ void wait_async_lds() {
#ifdef HAVE_ASYNC_LDS
#if __has_builtin(__builtin_amdgcn_s_wait_asynccnt)
  __builtin_amdgcn_s_wait_asynccnt(0);
#else
  asm volatile("s_wait_asynccnt 0" ::: "memory");
#endif
#endif
}

// ---------------------------------------------------------------------------
// K1 / K3 (templated): one block = one (batch, chunk, head).
// 128 threads = 4 wave32s. GEMM tile: 32x128 output per gate, 16x16 WMMA tiles.
// ---------------------------------------------------------------------------
template <bool PHASE1>
__global__ __launch_bounds__(128)
void rglru_chunk(const float* __restrict__ x,
                 const int*   __restrict__ segpos,
                 const float* __restrict__ w_x,
                 const float* __restrict__ b_x,
                 const float* __restrict__ w_a,
                 const float* __restrict__ b_a,
                 const float* __restrict__ a_param,
                 float* __restrict__ y,       // PHASE3 only
                 float* __restrict__ Asum,    // PHASE1 only
                 float* __restrict__ Bsum,    // PHASE1 only
                 const float* __restrict__ carry, // PHASE3 only
                 int L, int C)
{
  constexpr int XP = 128;  // f32 x pitch (floats)
  constexpr int BP = 136;  // bf16 pitch: keeps b128 frag loads 16B-aligned,
                           // breaks transpose-store bank conflicts
  constexpr int GP = 132;  // gate f32 pitch

  __shared__ __align__(16) unsigned char smem[128640];
  float*  xf  = (float*)(smem + 0);          // 32*128*4   = 16384
  __bf16* xbf = (__bf16*)(smem + 16384);     // 32*136*2   =  8704
  __bf16* wT[2];
  wT[0] = (__bf16*)(smem + 25088);           // 128*136*2  = 34816
  wT[1] = (__bf16*)(smem + 59904);           // 128*136*2  = 34816
  float* gate[2];
  gate[0] = (float*)(smem + 94720);          // 32*132*4   = 16896
  gate[1] = (float*)(smem + 111616);         // 32*132*4   = 16896
  int* seg = (int*)(smem + 128512);          // 32*4       =   128

  const int tid = threadIdx.x;
  const int bid = blockIdx.x;
  const int h   = bid % H_;
  const int c   = (bid / H_) % C;
  const int b   = bid / (H_ * C);
  const int t0  = c * CHUNK;

  // ---- async-load x chunk [CHUNK x 128] f32 into LDS; segment positions
  const float* xg = x + ((size_t)(b * L + t0)) * D_ + h * BW;
  for (int e = tid; e < CHUNK * (BW / 4); e += 128) {
    int row = e >> 5, c4 = e & 31;
    g2l_copy16(xf + row * XP + c4 * 4, xg + (size_t)row * D_ + c4 * 4);
  }
  if (tid < CHUNK) seg[tid] = segpos[b * L + t0 + tid];

  // ---- load weights for this head, transpose (N-major) + bf16-convert
  const float* wg[2] = { w_x + (size_t)h * BW * BW, w_a + (size_t)h * BW * BW };
  for (int g = 0; g < 2; ++g) {
    for (int e = tid; e < BW * (BW / 4); e += 128) {
      int i = e >> 5, c4 = e & 31;
      float4 v = *(const float4*)(wg[g] + (size_t)i * BW + c4 * 4);
      int j = c4 * 4;
      wT[g][(j + 0) * BP + i] = (__bf16)v.x;
      wT[g][(j + 1) * BP + i] = (__bf16)v.y;
      wT[g][(j + 2) * BP + i] = (__bf16)v.z;
      wT[g][(j + 3) * BP + i] = (__bf16)v.w;
    }
  }
  wait_async_lds();
  __syncthreads();

  // ---- bf16 copy of x tile (A operand)
  for (int e = tid; e < CHUNK * BW; e += 128) {
    int row = e >> 7, j = e & 127;
    xbf[row * BP + j] = (__bf16)xf[row * XP + j];
  }
  __syncthreads();

  // ---- gate GEMMs: gate[g] = x @ w[g], bf16 in / f32 accumulate
  {
    const int wave = tid >> 5;
    const int lane = tid & 31;
    const int l15  = lane & 15;
    const int kh   = lane >> 4;       // K-half selector per ISA layout
    const int mt   = wave & 1;        // M tile (rows mt*16..+15)
    const int ng   = wave >> 1;       // N group (4 tiles each)
    const __bf16* arow = xbf + (mt * 16 + l15) * BP;
    for (int g = 0; g < 2; ++g) {
      for (int nt = 0; nt < 4; ++nt) {
        const int n0 = (ng * 4 + nt) * 16;
        const __bf16* brow = wT[g] + (n0 + l15) * BP;
        v8f acc = {0.f,0.f,0.f,0.f,0.f,0.f,0.f,0.f};
#pragma unroll
        for (int k0 = 0; k0 < BW; k0 += 32) {
          FragBF fa, fb;
          // A 16x32 bf16: lanes0-15 K={8kh..8kh+7, 16+8kh..+7} (ISA 7.12.2)
          fa.h2[0] = *(const v8bf*)(arow + k0 + 8 * kh);
          fa.h2[1] = *(const v8bf*)(arow + k0 + 16 + 8 * kh);
          // B 32x16 bf16: lanes0-15 K={8kh..8kh+7, 16+8kh..+7} (same as A)
          fb.h2[0] = *(const v8bf*)(brow + k0 + 8 * kh);
          fb.h2[1] = *(const v8bf*)(brow + k0 + 16 + 8 * kh);
          acc = __builtin_amdgcn_wmma_f32_16x16x32_bf16(
              false, fa.v, false, fb.v, (short)0, acc, false, false);
        }
        // C/D layout: VGPR r -> row r + 8*kh, col = lane&15
#pragma unroll
        for (int r = 0; r < 8; ++r) {
          int m = mt * 16 + r + 8 * kh;
          gate[g][m * GP + n0 + l15] = acc[r];
        }
      }
    }
  }
  __syncthreads();

  // ---- per-channel gating + in-chunk sequential scan (thread = channel)
  {
    const int j = tid;                 // 0..127
    const int d = h * BW + j;
    const float c8  = 8.0f * log1pf(__expf(a_param[d]));  // 8*softplus
    const float bxj = b_x[h * BW + j];
    const float baj = b_a[h * BW + j];
    float hrun = 0.f, Ap = 1.f;
    if (!PHASE1) hrun = carry[((size_t)(b * C + c)) * D_ + d];
#pragma unroll 4
    for (int tt = 0; tt < CHUNK; ++tt) {
      float gx = gate[0][tt * GP + j] + bxj;
      float ga = gate[1][tt * GP + j] + baj;
      gx = 1.f / (1.f + __expf(-gx));
      ga = 1.f / (1.f + __expf(-ga));
      float la    = -c8 * ga;
      bool  rst   = (seg[tt] == 0);
      float a     = rst ? 0.f : __expf(la);
      float mult  = rst ? 1.f : sqrtf(1.0f - __expf(2.f * la) + 1e-6f);
      float xh    = xf[tt * XP + j] * gx * mult;
      hrun = fmaf(a, hrun, xh);
      if (PHASE1) Ap *= a;
      else y[((size_t)(b * L + t0 + tt)) * D_ + d] = hrun;
    }
    if (PHASE1) {
      size_t o = ((size_t)(b * C + c)) * D_ + d;
      Asum[o] = Ap;     // chunk transition: h_out = Ap*h_in + hrun
      Bsum[o] = hrun;
    }
  }
}

// ---------------------------------------------------------------------------
// K2: exclusive scan over chunk summaries, one thread per (b, d) channel.
// ---------------------------------------------------------------------------
__global__ __launch_bounds__(256)
void rglru_carry(const float* __restrict__ Asum,
                 const float* __restrict__ Bsum,
                 float* __restrict__ carry,
                 int C, int BD)
{
  int idx = blockIdx.x * blockDim.x + threadIdx.x;
  if (idx >= BD) return;
  int b = idx / D_, d = idx - b * D_;
  float h = 0.f;
  for (int cc = 0; cc < C; ++cc) {
    size_t o = ((size_t)(b * C + cc)) * D_ + d;
    carry[o] = h;                       // exclusive carry into chunk cc
    h = fmaf(Asum[o], h, Bsum[o]);
  }
}

// ---------------------------------------------------------------------------
extern "C" void kernel_launch(void* const* d_in, const int* in_sizes, int n_in,
                              void* d_out, int out_size, void* d_ws, size_t ws_size,
                              hipStream_t stream)
{
  (void)n_in; (void)out_size; (void)ws_size;
  const float* x       = (const float*)d_in[0];
  const int*   segpos  = (const int*)d_in[1];
  /* d_in[2] prev_h: unused by reference */
  const float* w_x     = (const float*)d_in[3];
  const float* b_x     = (const float*)d_in[4];
  const float* w_a     = (const float*)d_in[5];
  const float* b_a     = (const float*)d_in[6];
  const float* a_param = (const float*)d_in[7];
  float* y = (float*)d_out;

  const int BL = in_sizes[1];            // B*L
  const int Dd = in_sizes[0] / BL;       // 1024
  const int B  = in_sizes[2] / Dd;       // prev_h is [B, D]
  const int L  = BL / B;
  const int C  = L / CHUNK;

  const size_t n = (size_t)B * C * D_;
  float* Asum  = (float*)d_ws;
  float* Bsum  = Asum + n;
  float* carry = Bsum + n;

  dim3 blk(128);
  dim3 grd(B * C * H_);
  rglru_chunk<true><<<grd, blk, 0, stream>>>(x, segpos, w_x, b_x, w_a, b_a,
                                             a_param, nullptr, Asum, Bsum,
                                             nullptr, L, C);
  const int BD = B * D_;
  rglru_carry<<<(BD + 255) / 256, 256, 0, stream>>>(Asum, Bsum, carry, C, BD);
  rglru_chunk<false><<<grd, blk, 0, stream>>>(x, segpos, w_x, b_x, w_a, b_a,
                                              a_param, y, Asum, Bsum,
                                              carry, L, C);
}
